// SimpleAttention_9277129359737
// MI455X (gfx1250) — compile-verified, matrix operand order checked by reading, not run on hardware
//
#include <hip/hip_runtime.h>

// ---- CDNA5 WMMA types ----
typedef __attribute__((ext_vector_type(16))) __bf16 v16bf;
typedef __attribute__((ext_vector_type(8)))  __bf16 v8bf;
typedef __attribute__((ext_vector_type(8)))  float  v8f;

#define SEQ    2048
#define DIM    1024
#define BM     32          // query rows per workgroup
#define BN     64          // kv rows per iteration
#define DCHUNK 256         // d-columns staged per PV phase
#define Q_LD   1032        // padded bf16 Q row stride (16B-aligned)
#define VT_LD  72          // padded bf16 Vt row stride (16B-aligned)
#define S_LD   65          // padded fp32 S-tile row stride
#define NEGV   (-1e9f)

// 8 waves / 256 threads per block.
// QK role:  wave w -> S subtile (miq = w>>2, niq = w&3), 16x16 over K=1024
// PV role:  wave w -> O rows [16*(w&1), +16), cols { c*256 + (w>>1)*64 + [0,64) : c=0..3 }
__global__ __launch_bounds__(256) void fa_fwd_kernel(const float* __restrict__ X,
                                                     const int*   __restrict__ mask,
                                                     float*       __restrict__ out) {
    __shared__ __bf16 Qs[BM * Q_LD];        // 66048 B  bf16 Q tile (staged once)
    __shared__ __bf16 Vt[DCHUNK * VT_LD];   // 36864 B  transposed bf16 V chunk: Vt[d][n]
    __shared__ float  Sb[BM * S_LD];        //  8320 B  fp32 scores tile
    __shared__ __bf16 Pb[BM * BN];          //  4096 B  probs (bf16) for PV
    __shared__ float  red[BM * 8];          //  1024 B  softmax partial reductions
    __shared__ float  mrow[BM], lrow[BM], arow[BM];

    const int tid  = threadIdx.x;
    const int lane = tid & 31;
    const int wid  = tid >> 5;
    const int lh   = lane >> 4;   // lane half (0/1)
    const int ln   = lane & 15;

    const int b  = blockIdx.x / (SEQ / BM);
    const int q0 = (blockIdx.x % (SEQ / BM)) * BM;
    const float* Xb = X + (size_t)b * SEQ * DIM;
    const int*   mb = mask + b * SEQ;

    const int miq = wid >> 2, niq = wid & 3;   // QK tile coords
    const int mip = wid & 1,  ws  = wid >> 1;  // PV row-tile / 64-col subgroup

    // softmax role: 8 threads per row, 8 elements each
    const int sr = tid >> 3, ss = tid & 7;

    // 16 x v8f = 128 VGPRs of fp32 output accumulator per lane
    v8f acc[16];
#pragma unroll
    for (int nt = 0; nt < 16; ++nt)
#pragma unroll
        for (int j = 0; j < 8; ++j) acc[nt][j] = 0.0f;

    if (tid < BM) { mrow[tid] = -3.0e38f; lrow[tid] = 0.0f; }

    // ---------- stage Q tile once: bf16, row-major ----------
#pragma unroll 4
    for (int r = 0; r < BM; ++r) {
        const float4 f = *(const float4*)(Xb + (size_t)(q0 + r) * DIM + tid * 4);
        union { __bf16 e[4]; unsigned long long u; } pk;
        pk.e[0] = (__bf16)f.x; pk.e[1] = (__bf16)f.y;
        pk.e[2] = (__bf16)f.z; pk.e[3] = (__bf16)f.w;
        *(unsigned long long*)(Qs + r * Q_LD + tid * 4) = pk.u;
    }
    __syncthreads();

    // causal: only kv tiles with n0 <= q0+BM-1
    const int nsteps = (q0 + BM + BN - 1) / BN;
    const __bf16* qfrag = Qs + (miq * 16 + ln) * Q_LD + lh * 16;

    for (int step = 0; step < nsteps; ++step) {
        const int n0 = step * BN;

        // ---------- prefetch next KV tile toward this WGP ----------
        if (step + 1 < nsteps) {
            const float* nb = Xb + (size_t)(n0 + BN) * DIM;
#pragma unroll
            for (int it = 0; it < 8; ++it) {
                const int i   = it * 256 + tid;          // 2048 lines of 128B
                const int row = i >> 5;
                const int seg = i & 31;
                __builtin_prefetch(nb + (size_t)row * DIM + seg * 32, 0, 3);
            }
        }

        // ---------- QK: one 16x16 wmma tile per wave, K = 1024 ----------
        {
            v8f s;
#pragma unroll
            for (int j = 0; j < 8; ++j) s[j] = 0.0f;

            const float* kbase = Xb + (size_t)(n0 + niq * 16 + ln) * DIM + lh * 16;

            // software-pipelined K fragment loads (double buffer)
            float kf[2][16];
            {
                const float4* p = (const float4*)kbase;
                *(float4*)&kf[0][0]  = p[0]; *(float4*)&kf[0][4]  = p[1];
                *(float4*)&kf[0][8]  = p[2]; *(float4*)&kf[0][12] = p[3];
            }
#pragma unroll
            for (int i = 0; i < 32; ++i) {               // d0 = i*32
                const int cur = i & 1, nxt = cur ^ 1;
                if (i < 31) {
                    const float4* p = (const float4*)(kbase + (i + 1) * 32);
                    *(float4*)&kf[nxt][0]  = p[0]; *(float4*)&kf[nxt][4]  = p[1];
                    *(float4*)&kf[nxt][8]  = p[2]; *(float4*)&kf[nxt][12] = p[3];
                }
                union { v16bf v; v8bf h[2]; } aU;
                aU.h[0] = *(const v8bf*)(qfrag + i * 32);
                aU.h[1] = *(const v8bf*)(qfrag + i * 32 + 8);
                v16bf kb;
#pragma unroll
                for (int e = 0; e < 16; ++e) kb[e] = (__bf16)kf[cur][e];
                s = __builtin_amdgcn_wmma_f32_16x16x32_bf16(
                        false, aU.v, false, kb, (short)0, s, false, false);
            }
#pragma unroll
            for (int j = 0; j < 8; ++j)
                Sb[(miq * 16 + j + 8 * lh) * S_LD + niq * 16 + ln] = s[j];
        }
        __syncthreads();

        // ---------- online softmax (parallel: 8 threads x 8 elems per row) ----
        {
            float pmax = -3.0e38f;
#pragma unroll
            for (int e = 0; e < 8; ++e) {
                const int n = ss * 8 + e;
                const int k = n0 + n;
                float sv = Sb[sr * S_LD + n];
                if (k > q0 + sr) sv = NEGV;                          // causal
                else             sv += (1.0f - (float)mb[k]) * NEGV; // padding
                Sb[sr * S_LD + n] = sv;
                pmax = fmaxf(pmax, sv);
            }
            red[sr * 8 + ss] = pmax;
        }
        __syncthreads();
        if (tid < BM) {
            const int r = tid;
            float mt = mrow[r];
#pragma unroll
            for (int i = 0; i < 8; ++i) mt = fmaxf(mt, red[r * 8 + i]);
            arow[r] = __expf(mrow[r] - mt);   // alpha
            mrow[r] = mt;
        }
        __syncthreads();
        {
            const float mt = mrow[sr];
            float psum = 0.0f;
#pragma unroll
            for (int e = 0; e < 8; ++e) {
                const int n = ss * 8 + e;
                const float p = __expf(Sb[sr * S_LD + n] - mt);
                psum += p;
                Pb[sr * BN + n] = (__bf16)p;
            }
            red[sr * 8 + ss] = psum;
        }
        __syncthreads();
        if (tid < BM) {
            float lsum = 0.0f;
#pragma unroll
            for (int i = 0; i < 8; ++i) lsum += red[tid * 8 + i];
            lrow[tid] = lrow[tid] * arow[tid] + lsum;
        }

        // ---------- PV: rescale, then 4 staged d-chunk phases ----------
        union { v16bf v; v8bf h[2]; } aU[2];
#pragma unroll
        for (int ks = 0; ks < 2; ++ks) {
            const __bf16* ap = Pb + (mip * 16 + ln) * BN + ks * 32 + lh * 16;
            aU[ks].h[0] = *(const v8bf*)(ap);
            aU[ks].h[1] = *(const v8bf*)(ap + 8);
        }
#pragma unroll
        for (int j = 0; j < 8; ++j) {
            const float al = arow[mip * 16 + j + 8 * lh];
#pragma unroll
            for (int nt = 0; nt < 16; ++nt) acc[nt][j] *= al;
        }

        for (int c = 0; c < 4; ++c) {
            // stage Vt[d][n] = bf16(X[n0+n][c*256+d])
#pragma unroll
            for (int it = 0; it < 16; ++it) {
                const int i = it * 256 + tid;
                const int d = i & (DCHUNK - 1);
                const int n = (i >> 8) * 4;
                const float* src = Xb + (size_t)(n0 + n) * DIM + c * DCHUNK + d;
                union { __bf16 e[4]; unsigned long long u; } pk;
                pk.e[0] = (__bf16)src[0 * DIM];
                pk.e[1] = (__bf16)src[1 * DIM];
                pk.e[2] = (__bf16)src[2 * DIM];
                pk.e[3] = (__bf16)src[3 * DIM];
                *(unsigned long long*)(Vt + d * VT_LD + n) = pk.u;
            }
            __syncthreads();

            // 8 fragments (ks x ntl), software-pipelined double-buffered B
            union { v16bf v; v8bf h[2]; } bU[2];
            {
                const __bf16* vp0 = Vt + (ws * 64 + ln) * VT_LD + lh * 16;
                bU[0].h[0] = *(const v8bf*)(vp0);
                bU[0].h[1] = *(const v8bf*)(vp0 + 8);
            }
#pragma unroll
            for (int f = 0; f < 8; ++f) {
                const int ks  = f >> 2;
                const int ntl = f & 3;
                if (f < 7) {
                    const int nks  = (f + 1) >> 2;
                    const int nntl = (f + 1) & 3;
                    const __bf16* vp = Vt + (ws * 64 + nntl * 16 + ln) * VT_LD
                                          + nks * 32 + lh * 16;
                    bU[(f + 1) & 1].h[0] = *(const v8bf*)(vp);
                    bU[(f + 1) & 1].h[1] = *(const v8bf*)(vp + 8);
                }
                const int nt = c * 4 + ntl;
                acc[nt] = __builtin_amdgcn_wmma_f32_16x16x32_bf16(
                              false, aU[ks].v, false, bU[f & 1].v, (short)0,
                              acc[nt], false, false);
            }
            __syncthreads();
        }
    }

    // ---------- finalize: O = acc / l ----------
#pragma unroll
    for (int j = 0; j < 8; ++j) {
        const int r = mip * 16 + j + 8 * lh;
        const float inv = 1.0f / fmaxf(lrow[r], 1e-20f);
#pragma unroll
        for (int nt = 0; nt < 16; ++nt) {
            const int dcol = (nt >> 2) * 256 + ws * 64 + (nt & 3) * 16 + ln;
            out[((size_t)b * SEQ + q0 + r) * DIM + dcol] = acc[nt][j] * inv;
        }
    }
}

extern "C" void kernel_launch(void* const* d_in, const int* in_sizes, int n_in,
                              void* d_out, int out_size, void* d_ws, size_t ws_size,
                              hipStream_t stream) {
    const float* X    = (const float*)d_in[0];
    const int*   mask = (const int*)d_in[1];
    float*       out  = (float*)d_out;

    const int blocks = 4 * (SEQ / BM);   // 256 workgroups
    fa_fwd_kernel<<<blocks, 256, 0, stream>>>(X, mask, out);
}
